// SelfAttention_89017492177648
// MI455X (gfx1250) — compile-verified, matrix operand order checked by reading, not run on hardware
//
#include <hip/hip_runtime.h>

#define LQ 2048
#define DD 1024
#define NB 8

typedef __attribute__((ext_vector_type(16))) __bf16 v16bf;
typedef __attribute__((ext_vector_type(8)))  float  v8f;

// ---- helpers -------------------------------------------------------------

__device__ __forceinline__ unsigned short f2bf(float f) {
  __bf16 h = (__bf16)f;  // RNE, lowers to HW cvt
  return __builtin_bit_cast(unsigned short, h);
}
__device__ __forceinline__ unsigned int pack2(float lo, float hi) {
  return (unsigned int)f2bf(lo) | ((unsigned int)f2bf(hi) << 16);
}
__device__ __forceinline__ float bf2f(unsigned short u) {
  return __builtin_bit_cast(float, (unsigned int)u << 16);
}

// Load one 16x32 bf16 WMMA operand fragment for this lane from an LDS row.
// 16-bit operand layout: lanes 0-15 (h=0): regs0-3 K=0..7, regs4-7 K=16..23;
// lanes 16-31 (h=1): K=8..15 and K=24..31  ->  two ds_load_b128 per lane.
__device__ __forceinline__ v16bf ld_frag(const unsigned short* p, int h) {
  union { v16bf v; uint4 q[2]; } u;
  u.q[0] = *(const uint4*)(p + 8 * h);
  u.q[1] = *(const uint4*)(p + 16 + 8 * h);
  return u.v;
}

__device__ __forceinline__ v8f wmma_bf16(v16bf a, v16bf b, v8f c) {
  return __builtin_amdgcn_wmma_f32_16x16x32_bf16(false, a, false, b, (short)0, c,
                                                 false, false);
}

// ---- kernel P: one-off split X (f32) -> Xhi + Xlo (bf16 pair) ------------
// x = hi + lo to ~17 mantissa bits: hi = bf16(x), lo = bf16(x - hi).

__global__ __launch_bounds__(256) void k_prep(const float* __restrict__ X,
                                              unsigned short* __restrict__ Xhi,
                                              unsigned short* __restrict__ Xlo) {
  const size_t i0 = ((size_t)blockIdx.x * 256 + threadIdx.x) * 8;
  const float4* s = (const float4*)(X + i0);
  float4 f0 = s[0], f1 = s[1];
  float x[8] = {f0.x, f0.y, f0.z, f0.w, f1.x, f1.y, f1.z, f1.w};
  unsigned short hi[8], lo[8];
#pragma unroll
  for (int i = 0; i < 8; ++i) {
    hi[i] = f2bf(x[i]);
    lo[i] = f2bf(x[i] - bf2f(hi[i]));
  }
  uint4 ph, pl;
  ph.x = hi[0] | ((unsigned)hi[1] << 16); ph.y = hi[2] | ((unsigned)hi[3] << 16);
  ph.z = hi[4] | ((unsigned)hi[5] << 16); ph.w = hi[6] | ((unsigned)hi[7] << 16);
  pl.x = lo[0] | ((unsigned)lo[1] << 16); pl.y = lo[2] | ((unsigned)lo[3] << 16);
  pl.z = lo[4] | ((unsigned)lo[5] << 16); pl.w = lo[6] | ((unsigned)lo[7] << 16);
  *(uint4*)(Xhi + i0) = ph;
  *(uint4*)(Xlo + i0) = pl;
}

// ---- kernel 0: zero the column-sum accumulator ---------------------------

__global__ void k_zero(float* __restrict__ p, int n) {
  int i = blockIdx.x * blockDim.x + threadIdx.x;
  if (i < n) p[i] = 0.0f;
}

// ---- kernel 1: per-row max of S = X X^T (== per-column max, S symmetric) -
// Block tile 64x128, BK=32, 8 waves 2x4, wave 32x32, double-buffered LDS.
// bf16-hi only: M just needs to be within ~87 of the true max (it is, by far)
// because the same M is used for exp() and the normalizing sum.

__global__ __launch_bounds__(256) void k_rowmax(const unsigned short* __restrict__ Xbf,
                                                float* __restrict__ Mmax) {
  __shared__ unsigned short As[2][64][40];
  __shared__ unsigned short Bs[2][128][40];
  __shared__ float red[64][4];

  const int t = threadIdx.x;
  const int wid = t >> 5, lane = t & 31, ln = lane & 15, h = lane >> 4;
  const int wm = wid & 1, wn = wid >> 1;
  const int b = blockIdx.y, l0 = blockIdx.x * 64;
  const unsigned short* Xb = Xbf + (size_t)b * LQ * DD;

  const int arow = t >> 2, ac = (t & 3) * 8;   // A stage: 64x32, 1 uint4/thread
  const int brow = t >> 1, bc = (t & 1) * 16;  // B stage: 128x32, 2 uint4/thread

  auto stage = [&](int s) {  // s = mt*32 + kt, buffer s&1
    const int m0 = (s >> 5) * 128, d0 = (s & 31) * 32, buf = s & 1;
    *(uint4*)&As[buf][arow][ac] =
        *(const uint4*)(Xb + (size_t)(l0 + arow) * DD + d0 + ac);
    const uint4* bsrc = (const uint4*)(Xb + (size_t)(m0 + brow) * DD + d0 + bc);
    *(uint4*)&Bs[buf][brow][bc] = bsrc[0];
    *(uint4*)&Bs[buf][brow][bc + 8] = bsrc[1];
  };

  float rmax[16];
#pragma unroll
  for (int i = 0; i < 16; ++i) rmax[i] = -3.0e38f;

  stage(0);
  for (int mt = 0; mt < LQ / 128; ++mt) {
    v8f acc[2][2];
#pragma unroll
    for (int i = 0; i < 2; ++i)
#pragma unroll
      for (int j = 0; j < 2; ++j)
#pragma unroll
        for (int r = 0; r < 8; ++r) acc[i][j][r] = 0.0f;

    for (int kt = 0; kt < DD / 32; ++kt) {
      const int s = mt * (DD / 32) + kt, buf = s & 1;
      __syncthreads();
      if (s + 1 < (LQ / 128) * (DD / 32)) stage(s + 1);
      v16bf af[2], bfv[2];
#pragma unroll
      for (int fm = 0; fm < 2; ++fm)
        af[fm] = ld_frag(&As[buf][wm * 32 + fm * 16 + ln][0], h);
#pragma unroll
      for (int fn = 0; fn < 2; ++fn)
        bfv[fn] = ld_frag(&Bs[buf][wn * 32 + fn * 16 + ln][0], h);
#pragma unroll
      for (int fm = 0; fm < 2; ++fm)
#pragma unroll
        for (int fn = 0; fn < 2; ++fn)
          acc[fm][fn] = wmma_bf16(af[fm], bfv[fn], acc[fm][fn]);
    }
#pragma unroll
    for (int fm = 0; fm < 2; ++fm)
#pragma unroll
      for (int r = 0; r < 8; ++r) {
        float v = fmaxf(acc[fm][0][r], acc[fm][1][r]);
        rmax[fm * 8 + r] = fmaxf(rmax[fm * 8 + r], v);
      }
  }

  // reduce over the 16 columns held by this half-wave
#pragma unroll
  for (int i = 0; i < 16; ++i) {
    float v = rmax[i];
#pragma unroll
    for (int off = 1; off < 16; off <<= 1) v = fmaxf(v, __shfl_xor(v, off, 32));
    rmax[i] = v;
  }
  __syncthreads();
  if (ln == 0) {
#pragma unroll
    for (int fm = 0; fm < 2; ++fm)
#pragma unroll
      for (int r = 0; r < 8; ++r)
        red[wm * 32 + fm * 16 + 8 * h + r][wn] = rmax[fm * 8 + r];
  }
  __syncthreads();
  if (t < 64) {
    float v = fmaxf(fmaxf(red[t][0], red[t][1]), fmaxf(red[t][2], red[t][3]));
    Mmax[b * LQ + l0 + t] = v;
  }
}

// ---- kernel 2: E = exp(S - M[col]) (bf16) + column sums ------------------
// Split-precision S: S = hi*hi^T + hi*lo^T + lo*hi^T (abs err ~1e-3).
// Implemented as a virtual K of 3*1024: kt in [0,96), term = kt/32 selects
// the (A,B) source planes. Block tile 128x128, BK=32, waves 4x2 (2x4 frags).

__global__ __launch_bounds__(256) void k_expsum(const unsigned short* __restrict__ Xhi,
                                                const unsigned short* __restrict__ Xlo,
                                                const float* __restrict__ Mmax,
                                                unsigned short* __restrict__ E,
                                                float* __restrict__ Ssum) {
  __shared__ unsigned short As[2][128][40];
  __shared__ unsigned short Bs[2][128][40];

  const int t = threadIdx.x;
  const int wid = t >> 5, lane = t & 31, ln = lane & 15, h = lane >> 4;
  const int wm = wid >> 1, wn = wid & 1;
  const int b = blockIdx.z, l0 = blockIdx.x * 128, m0 = blockIdx.y * 128;
  const unsigned short* XhiB = Xhi + (size_t)b * LQ * DD;
  const unsigned short* XloB = Xlo + (size_t)b * LQ * DD;

  const int srow = t >> 1, sc = (t & 1) * 16;  // 128x32, 2 uint4/thread each

  auto stage = [&](int kt) {
    const int term = kt >> 5;          // 0: hi*hi, 1: hi*lo, 2: lo*hi
    const int d0 = (kt & 31) * 32, buf = kt & 1;
    const unsigned short* ap = (term == 2) ? XloB : XhiB;  // A plane (rows l)
    const unsigned short* bp = (term == 1) ? XloB : XhiB;  // B plane (rows m)
    const uint4* asrc = (const uint4*)(ap + (size_t)(l0 + srow) * DD + d0 + sc);
    *(uint4*)&As[buf][srow][sc] = asrc[0];
    *(uint4*)&As[buf][srow][sc + 8] = asrc[1];
    const uint4* bsrc = (const uint4*)(bp + (size_t)(m0 + srow) * DD + d0 + sc);
    *(uint4*)&Bs[buf][srow][sc] = bsrc[0];
    *(uint4*)&Bs[buf][srow][sc + 8] = bsrc[1];
  };

  v8f acc[2][4];
#pragma unroll
  for (int i = 0; i < 2; ++i)
#pragma unroll
    for (int j = 0; j < 4; ++j)
#pragma unroll
      for (int r = 0; r < 8; ++r) acc[i][j][r] = 0.0f;

  stage(0);
  const int NK = 3 * (DD / 32);
  for (int kt = 0; kt < NK; ++kt) {
    const int buf = kt & 1;
    __syncthreads();
    if (kt + 1 < NK) stage(kt + 1);
    v16bf af[2], bfv[4];
#pragma unroll
    for (int fm = 0; fm < 2; ++fm)
      af[fm] = ld_frag(&As[buf][wm * 32 + fm * 16 + ln][0], h);
#pragma unroll
    for (int fn = 0; fn < 4; ++fn)
      bfv[fn] = ld_frag(&Bs[buf][wn * 64 + fn * 16 + ln][0], h);
#pragma unroll
    for (int fm = 0; fm < 2; ++fm)
#pragma unroll
      for (int fn = 0; fn < 4; ++fn)
        acc[fm][fn] = wmma_bf16(af[fm], bfv[fn], acc[fm][fn]);
  }

  // epilogue: exp, store E bf16, accumulate column sums
#pragma unroll
  for (int fn = 0; fn < 4; ++fn) {
    const int col = m0 + wn * 64 + fn * 16 + ln;
    const float mx = Mmax[b * LQ + col];
    float cs = 0.0f;
#pragma unroll
    for (int fm = 0; fm < 2; ++fm) {
      const int row0 = l0 + wm * 32 + fm * 16 + 8 * h;
      unsigned short* ep = E + ((size_t)b * LQ + row0) * LQ + col;
#pragma unroll
      for (int r = 0; r < 8; ++r) {
        float e = __expf(acc[fm][fn][r] - mx);  // M within ~0.1 of max: e <= ~1.1
        ep[(size_t)r * LQ] = f2bf(e);
        cs += e;
      }
    }
    cs += __shfl_xor(cs, 16, 32);  // combine row-halves (same column)
    if (h == 0) atomicAdd(&Ssum[b * LQ + col], cs);
  }
}

// ---- kernel T: one-off XwT[b][d][m] = bf16( X[b][m][d] / Ssum[b][m] ) ----
// 64x64 tiles transposed through LDS.

__global__ __launch_bounds__(256) void k_scaleT(const float* __restrict__ X,
                                                const float* __restrict__ Ssum,
                                                unsigned short* __restrict__ XwT) {
  __shared__ float T[64][68];
  const int t = threadIdx.x;
  const int b = blockIdx.z, m0 = blockIdx.x * 64, d0 = blockIdx.y * 64;
  const float* Xb = X + (size_t)b * LQ * DD;

  {  // load rows of X, scale by 1/colsum, store row-major into LDS
    const int mr = t >> 2, dc = (t & 3) * 16;
    const float sw = 1.0f / Ssum[b * LQ + m0 + mr];
    const float4* s = (const float4*)(Xb + (size_t)(m0 + mr) * DD + d0 + dc);
#pragma unroll
    for (int q = 0; q < 4; ++q) {
      float4 f = s[q];
      float4 g; g.x = f.x * sw; g.y = f.y * sw; g.z = f.z * sw; g.w = f.w * sw;
      *(float4*)&T[mr][dc + 4 * q] = g;
    }
  }
  __syncthreads();
  {  // read transposed, convert to bf16, store rows of XwT
    const int dr = t >> 2, mc = (t & 3) * 16;
    uint4 p0, p1;
    p0.x = pack2(T[mc + 0][dr],  T[mc + 1][dr]);
    p0.y = pack2(T[mc + 2][dr],  T[mc + 3][dr]);
    p0.z = pack2(T[mc + 4][dr],  T[mc + 5][dr]);
    p0.w = pack2(T[mc + 6][dr],  T[mc + 7][dr]);
    p1.x = pack2(T[mc + 8][dr],  T[mc + 9][dr]);
    p1.y = pack2(T[mc + 10][dr], T[mc + 11][dr]);
    p1.z = pack2(T[mc + 12][dr], T[mc + 13][dr]);
    p1.w = pack2(T[mc + 14][dr], T[mc + 15][dr]);
    unsigned short* o = XwT + ((size_t)b * DD + d0 + dr) * LQ + m0 + mc;
    *(uint4*)o = p0;
    *(uint4*)(o + 8) = p1;
  }
}

// ---- kernel 3: out = E @ XwT^T (both operands already bf16) --------------
// Block tile 128x128, BK=32, 8 waves 4x2, wave 32x64, double-buffered LDS.

__global__ __launch_bounds__(256) void k_out(const unsigned short* __restrict__ E,
                                             const unsigned short* __restrict__ XwT,
                                             float* __restrict__ Out) {
  __shared__ unsigned short As[2][128][40];
  __shared__ unsigned short Bs[2][128][40];

  const int t = threadIdx.x;
  const int wid = t >> 5, lane = t & 31, ln = lane & 15, h = lane >> 4;
  const int wm = wid >> 1, wn = wid & 1;
  const int b = blockIdx.z, l0 = blockIdx.x * 128, n0 = blockIdx.y * 128;
  const unsigned short* Eb = E + (size_t)b * LQ * LQ;
  const unsigned short* Wb = XwT + (size_t)b * DD * LQ;

  const int srow = t >> 1, sc = (t & 1) * 16;

  auto stage = [&](int kt) {
    const int m0 = kt * 32, buf = kt & 1;
    const uint4* asrc = (const uint4*)(Eb + (size_t)(l0 + srow) * LQ + m0 + sc);
    *(uint4*)&As[buf][srow][sc] = asrc[0];
    *(uint4*)&As[buf][srow][sc + 8] = asrc[1];
    const uint4* bsrc = (const uint4*)(Wb + (size_t)(n0 + srow) * LQ + m0 + sc);
    *(uint4*)&Bs[buf][srow][sc] = bsrc[0];
    *(uint4*)&Bs[buf][srow][sc + 8] = bsrc[1];
  };

  v8f acc[2][4];
#pragma unroll
  for (int i = 0; i < 2; ++i)
#pragma unroll
    for (int j = 0; j < 4; ++j)
#pragma unroll
      for (int r = 0; r < 8; ++r) acc[i][j][r] = 0.0f;

  stage(0);
  for (int kt = 0; kt < LQ / 32; ++kt) {
    const int buf = kt & 1;
    __syncthreads();
    if (kt + 1 < LQ / 32) stage(kt + 1);
    v16bf af[2], bfv[4];
#pragma unroll
    for (int fm = 0; fm < 2; ++fm)
      af[fm] = ld_frag(&As[buf][wm * 32 + fm * 16 + ln][0], h);
#pragma unroll
    for (int fn = 0; fn < 4; ++fn)
      bfv[fn] = ld_frag(&Bs[buf][wn * 64 + fn * 16 + ln][0], h);
#pragma unroll
    for (int fm = 0; fm < 2; ++fm)
#pragma unroll
      for (int fn = 0; fn < 4; ++fn)
        acc[fm][fn] = wmma_bf16(af[fm], bfv[fn], acc[fm][fn]);
  }

  // epilogue: f32 stores, 16 lanes -> 64B contiguous per fragment row
#pragma unroll
  for (int fm = 0; fm < 2; ++fm) {
    const int row0 = l0 + wm * 32 + fm * 16 + 8 * h;
#pragma unroll
    for (int fn = 0; fn < 4; ++fn) {
      const int col = n0 + wn * 64 + fn * 16 + ln;
      float* op = Out + ((size_t)b * LQ + row0) * DD + col;
#pragma unroll
      for (int r = 0; r < 8; ++r) op[(size_t)r * DD] = acc[fm][fn][r];
    }
  }
}

// ---- host-side launcher --------------------------------------------------
// ws layout: [0,64K) col max | [64K,128K) col sums | [128K, +64M) E bf16
//            | [+64M, +96M) Xhi | [+96M, +128M) Xlo | [+128M, +160M) XwT
// Requires ws_size >= 160MB + 128KB.

extern "C" void kernel_launch(void* const* d_in, const int* in_sizes, int n_in,
                              void* d_out, int out_size, void* d_ws, size_t ws_size,
                              hipStream_t stream) {
  (void)in_sizes; (void)n_in; (void)out_size; (void)ws_size;
  const float* X = (const float*)d_in[0];
  float* Out = (float*)d_out;

  char* wsc = (char*)d_ws;
  float* Mmax = (float*)wsc;
  float* Ssum = (float*)(wsc + (64 << 10));
  unsigned short* E   = (unsigned short*)(wsc + (128 << 10));
  unsigned short* Xhi = (unsigned short*)(wsc + (128 << 10) + ((size_t)64 << 20));
  unsigned short* Xlo = (unsigned short*)(wsc + (128 << 10) + ((size_t)96 << 20));
  unsigned short* XwT = (unsigned short*)(wsc + (128 << 10) + ((size_t)128 << 20));

  const size_t nelem = (size_t)NB * LQ * DD;
  k_prep<<<(unsigned)(nelem / (256 * 8)), 256, 0, stream>>>(X, Xhi, Xlo);
  k_zero<<<(NB * LQ + 255) / 256, 256, 0, stream>>>(Ssum, NB * LQ);
  k_rowmax<<<dim3(LQ / 64, NB), 256, 0, stream>>>(Xhi, Mmax);
  k_expsum<<<dim3(LQ / 128, LQ / 128, NB), 256, 0, stream>>>(Xhi, Xlo, Mmax, E, Ssum);
  k_scaleT<<<dim3(LQ / 64, DD / 64, NB), 256, 0, stream>>>(X, Ssum, XwT);
  k_out<<<dim3(LQ / 128, DD / 128, NB), 256, 0, stream>>>(E, XwT, Out);
}
